// MyRanker_5961414607255
// MI455X (gfx1250) — compile-verified, matrix operand order checked by reading, not run on hardware
//
#include <hip/hip_runtime.h>
#include <hip/hip_bf16.h>
#include <math.h>

// ---------------------------------------------------------------------------
// Problem constants (match reference)
// ---------------------------------------------------------------------------
#define DD 1024
#define HH 512
#define NHEAD 8
#define DHEAD 128
#define BB 16
#define SS 1024
#define QQ 128
#define TT 16
#define LTT 8
#define CC 64
#define LCC 16

// ---------------------------------------------------------------------------
// WMMA types (CDNA5 gfx1250, wave32)
// ---------------------------------------------------------------------------
typedef __attribute__((ext_vector_type(16))) __bf16 v16bf;
typedef __attribute__((ext_vector_type(8)))  __bf16 v8bf;
typedef __attribute__((ext_vector_type(8)))  float  v8f;

#define GBM 128
#define GBN 128
#define GBK 32

__device__ __forceinline__ float sigf(float x) { return 1.f / (1.f + __expf(-x)); }

// Convert a register-held 16-float half-row to bf16 and store into the LDS tile.
__device__ __forceinline__ void store_tile(__bf16 (&dst)[GBM][GBK],
                                           const float4 (&v)[4], int r, int c0) {
#pragma unroll
  for (int j = 0; j < 4; ++j) {
    dst[r][c0 + 4 * j + 0] = (__bf16)v[j].x;
    dst[r][c0 + 4 * j + 1] = (__bf16)v[j].y;
    dst[r][c0 + 4 * j + 2] = (__bf16)v[j].z;
    dst[r][c0 + 4 * j + 3] = (__bf16)v[j].w;
  }
}

// A/B fragment per ISA 16-bit 16x32 layout: lane holds row (lane&15); K chunks
// {b..b+7} and {b+16..b+23} with b = 8*(lane>=16). Two 16B LDS loads.
__device__ __forceinline__ v16bf load_frag(const __bf16 (&t)[GBM][GBK], int row, int hi) {
  const __bf16* p = &t[row][hi << 3];
  v8bf lo  = *(const v8bf*)p;
  v8bf hi8 = *(const v8bf*)(p + 16);
  v16bf f;
#pragma unroll
  for (int j = 0; j < 8; ++j) { f[j] = lo[j]; f[j + 8] = hi8[j]; }
  return f;
}

// C[M,N] = act( A1[M,K1] @ W1[N,K1]^T  (+ A2[M,K2] @ W2[N,K2]^T)  + bias[N] )
// Dual-A form implements the LSTM gate GEMM (x @ Wih^T + h @ Whh^T + b) fused.
// grid = (M/128, N/128), block = 256 (8 waves: 4 in M x 2 in N, 2x4 frags each).
// Ping-pong double-buffered LDS: one barrier per K-chunk; global loads for
// chunk i+1 issued before the WMMAs of chunk i; global_prefetch for chunk i+2.
__global__ __launch_bounds__(256) void gemm_bf16_wmma(
    const float* __restrict__ A1, int lda1, int K1,
    const float* __restrict__ A2, int lda2, int K2,
    const float* __restrict__ W1, const float* __restrict__ W2,
    const float* __restrict__ bias,
    float* __restrict__ C, int ldc, int act) {
  __shared__ __bf16 As[2][GBM][GBK];
  __shared__ __bf16 Bs[2][GBN][GBK];
  const int tid  = threadIdx.x;
  const int lane = tid & 31;
  const int wave = tid >> 5;
  const int wm = wave >> 1;       // 0..3
  const int wn = wave & 1;        // 0..1
  const int hi = lane >> 4;       // 0/1
  const int lr = lane & 15;
  const int bm = blockIdx.x * GBM;
  const int bn = blockIdx.y * GBN;

  const int r  = tid >> 1;        // 0..127 : tile row this thread stages
  const int c0 = (tid & 1) << 4;  // 0 / 16 : first of 16 staged columns

  const int n1 = K1 >> 5;
  const int n2 = A2 ? (K2 >> 5) : 0;
  const int n  = n1 + n2;

  auto addrA = [&](int ci) -> const float* {
    return (ci < n1) ? A1 + (size_t)(bm + r) * lda1 + (ci << 5) + c0
                     : A2 + (size_t)(bm + r) * lda2 + ((ci - n1) << 5) + c0;
  };
  auto addrB = [&](int ci) -> const float* {
    return (ci < n1) ? W1 + (size_t)(bn + r) * K1 + (ci << 5) + c0
                     : W2 + (size_t)(bn + r) * K2 + ((ci - n1) << 5) + c0;
  };

  v8f acc[2][4] = {};
  float4 ra[4], rb[4];

  // Preload chunk 0 and stage into buffer 0.
  {
    const float4* pa = (const float4*)addrA(0);
    const float4* pb = (const float4*)addrB(0);
#pragma unroll
    for (int j = 0; j < 4; ++j) { ra[j] = pa[j]; rb[j] = pb[j]; }
  }
  store_tile(As[0], ra, r, c0);
  store_tile(Bs[0], rb, r, c0);
  __syncthreads();

  int cur = 0;
  for (int ci = 0; ci < n; ++ci) {
    // Issue global loads for the next chunk early (latency overlaps the WMMAs).
    if (ci + 1 < n) {
      const float4* pa = (const float4*)addrA(ci + 1);
      const float4* pb = (const float4*)addrB(ci + 1);
#pragma unroll
      for (int j = 0; j < 4; ++j) { ra[j] = pa[j]; rb[j] = pb[j]; }
    }
    // Pull chunk i+2 toward L2/WGP$ (gfx1250 global_prefetch_b8).
    if (ci + 2 < n) {
      __builtin_prefetch(addrA(ci + 2), 0, 3);
      __builtin_prefetch(addrB(ci + 2), 0, 3);
    }

    // Compute on the current LDS buffer.
    v16bf aF[2], bF[4];
#pragma unroll
    for (int mi = 0; mi < 2; ++mi) aF[mi] = load_frag(As[cur], wm * 32 + mi * 16 + lr, hi);
#pragma unroll
    for (int ni = 0; ni < 4; ++ni) bF[ni] = load_frag(Bs[cur], wn * 64 + ni * 16 + lr, hi);
#pragma unroll
    for (int mi = 0; mi < 2; ++mi)
#pragma unroll
      for (int ni = 0; ni < 4; ++ni)
        acc[mi][ni] = __builtin_amdgcn_wmma_f32_16x16x32_bf16(
            false, aF[mi], false, bF[ni], (short)0, acc[mi][ni], false, false);

    // Stage the prefetched chunk into the alternate buffer; single barrier.
    if (ci + 1 < n) {
      store_tile(As[cur ^ 1], ra, r, c0);
      store_tile(Bs[cur ^ 1], rb, r, c0);
      __syncthreads();
      cur ^= 1;
    }
  }

  // Epilogue per ISA C/D layout: lane -> N = (lane&15); VGPR r -> M = r + 8*(lane>=16)
#pragma unroll
  for (int mi = 0; mi < 2; ++mi) {
#pragma unroll
    for (int ni = 0; ni < 4; ++ni) {
      const int gn  = bn + wn * 64 + ni * 16 + lr;
      const float bv = bias ? bias[gn] : 0.f;
      const int gm0 = bm + wm * 32 + mi * 16 + hi * 8;
#pragma unroll
      for (int q = 0; q < 8; ++q) {
        float v = acc[mi][ni][q] + bv;
        if (act == 1) v = (v > 0.f) ? v : 0.01f * v;   // leaky_relu(0.01)
        C[(size_t)(gm0 + q) * ldc + gn] = v;
      }
    }
  }
}

// ---------------------------------------------------------------------------
// Pointwise LSTM cell: gates[ns,2048] (i,f,g,o) -> update c,h; scatter h into
// the layer-output sequence buffer; optionally record final hidden (pool).
// ---------------------------------------------------------------------------
__global__ void lstm_point(const float* __restrict__ gates,
                           float* __restrict__ c, float* __restrict__ h,
                           float* __restrict__ Y, long ystride,
                           float* __restrict__ pooled, int ns) {
  int idx = blockIdx.x * 256 + threadIdx.x;
  if (idx >= ns * HH) return;
  int r = idx >> 9, j = idx & 511;
  const float* g = gates + (size_t)r * 2048;
  float iv = sigf(g[j]);
  float fv = sigf(g[512 + j]);
  float gv = tanhf(g[1024 + j]);
  float ov = sigf(g[1536 + j]);
  float cv = fv * c[idx] + iv * gv;
  c[idx] = cv;
  float hv = ov * tanhf(cv);
  h[idx] = hv;
  Y[(size_t)r * ystride + j] = hv;
  if (pooled) pooled[(size_t)r * DD + j] = hv;
}

// Gather seq rows by index: out[row,:1024] = seq[b*S + ids[row], :1024]
__global__ void gather_rows(const float* __restrict__ seq, const int* __restrict__ ids,
                            float* __restrict__ out, int rows_per_b) {
  int row = blockIdx.x;
  int b = row / rows_per_b;
  int src = b * SS + ids[row];
  const float4* s = (const float4*)(seq + (size_t)src * DD);
  float4* d = (float4*)(out + (size_t)row * DD);
  d[threadIdx.x] = s[threadIdx.x];           // 256 threads x float4 = 1024 floats
}

// Per (query-row, head) softmax attention. keys for row r start at (r/qdiv)*kmul.
__global__ void attn_kernel(const float* __restrict__ qp, const float* __restrict__ kp,
                            const float* __restrict__ vp, float* __restrict__ out,
                            int qdiv, int kmul, int nk) {
  __shared__ float red[DHEAD];
  __shared__ float sc[128];
  const int qrow = blockIdx.x >> 3;
  const int head = blockIdx.x & 7;
  const int d = threadIdx.x;                 // 0..127
  const int kbase = (qrow / qdiv) * kmul;
  const float scale = 0.08838834764831845f;  // 1/sqrt(128)
  float qv = qp[(size_t)qrow * DD + head * DHEAD + d];
  for (int k = 0; k < nk; ++k) {
    red[d] = qv * kp[(size_t)(kbase + k) * DD + head * DHEAD + d];
    __syncthreads();
    for (int s = 64; s > 0; s >>= 1) { if (d < s) red[d] += red[d + s]; __syncthreads(); }
    if (d == 0) sc[k] = red[0] * scale;
    __syncthreads();
  }
  red[d] = (d < nk) ? sc[d] : -3.4e38f;
  __syncthreads();
  for (int s = 64; s > 0; s >>= 1) { if (d < s) red[d] = fmaxf(red[d], red[d + s]); __syncthreads(); }
  float mx = red[0];
  __syncthreads();
  float e = (d < nk) ? __expf(sc[d] - mx) : 0.f;
  red[d] = e;
  __syncthreads();
  for (int s = 64; s > 0; s >>= 1) { if (d < s) red[d] += red[d + s]; __syncthreads(); }
  float inv = 1.f / red[0];
  __syncthreads();
  if (d < nk) sc[d] = e * inv;
  __syncthreads();
  float o = 0.f;
  for (int k = 0; k < nk; ++k) o += sc[k] * vp[(size_t)(kbase + k) * DD + head * DHEAD + d];
  out[(size_t)qrow * DD + head * DHEAD + d] = o;
}

// o = l2norm(a + b) per 1024-wide row
__global__ void resid_l2norm(const float* __restrict__ a, const float* __restrict__ b,
                             float* __restrict__ o) {
  __shared__ float red[256];
  int row = blockIdx.x, t = threadIdx.x;
  float vals[4]; float ss = 0.f;
#pragma unroll
  for (int i = 0; i < 4; ++i) {
    int j = t + i * 256;
    float v = a[(size_t)row * DD + j] + b[(size_t)row * DD + j];
    vals[i] = v; ss += v * v;
  }
  red[t] = ss; __syncthreads();
  for (int s = 128; s > 0; s >>= 1) { if (t < s) red[t] += red[t + s]; __syncthreads(); }
  float den = fmaxf(sqrtf(red[0]), 1e-12f);
#pragma unroll
  for (int i = 0; i < 4; ++i) o[(size_t)row * DD + t + i * 256] = vals[i] / den;
}

// logits[row, 0..1] = h1[row,:256] @ w2[2,256]^T + b2
__global__ void cls2_kernel(const float* __restrict__ h1, const float* __restrict__ w2,
                            const float* __restrict__ b2, float* __restrict__ out) {
  __shared__ float red[256];
  int row = blockIdx.x, t = threadIdx.x;
  float hv = h1[(size_t)row * 256 + t];
#pragma unroll
  for (int c = 0; c < 2; ++c) {
    red[t] = hv * w2[c * 256 + t];
    __syncthreads();
    for (int s = 128; s > 0; s >>= 1) { if (t < s) red[t] += red[t + s]; __syncthreads(); }
    if (t == 0) out[row * 2 + c] = red[0] + b2[c];
    __syncthreads();
  }
}

// ---------------------------------------------------------------------------
// Host orchestration
// ---------------------------------------------------------------------------
extern "C" void kernel_launch(void* const* d_in, const int* in_sizes, int n_in,
                              void* d_out, int out_size, void* d_ws, size_t ws_size,
                              hipStream_t stream) {
  const float* seq     = (const float*)d_in[0];
  const int*   q_ids   = (const int*)d_in[1];
  const int*   t_ids   = (const int*)d_in[2];
  const int*   c_ids   = (const int*)d_in[3];
  const float* tWih    = (const float*)d_in[4];
  const float* tWhh    = (const float*)d_in[5];
  const float* tBb     = (const float*)d_in[6];
  const float* cWih    = (const float*)d_in[7];
  const float* cWhh    = (const float*)d_in[8];
  const float* cBb     = (const float*)d_in[9];
  const float* t_pw    = (const float*)d_in[10];
  const float* t_pb    = (const float*)d_in[11];
  const float* c_pw    = (const float*)d_in[12];
  const float* c_pb    = (const float*)d_in[13];
  const float* tc_iw   = (const float*)d_in[14];
  const float* tc_ib   = (const float*)d_in[15];
  const float* tc_ow   = (const float*)d_in[16];
  const float* tc_ob   = (const float*)d_in[17];
  const float* tq_iw   = (const float*)d_in[18];
  const float* tq_ib   = (const float*)d_in[19];
  const float* tq_ow   = (const float*)d_in[20];
  const float* tq_ob   = (const float*)d_in[21];
  const float* cq_iw   = (const float*)d_in[22];
  const float* cq_ib   = (const float*)d_in[23];
  const float* cq_ow   = (const float*)d_in[24];
  const float* cq_ob   = (const float*)d_in[25];
  const float* t_w1    = (const float*)d_in[26];
  const float* t_b1    = (const float*)d_in[27];
  const float* t_w2    = (const float*)d_in[28];
  const float* t_b2    = (const float*)d_in[29];
  const float* c_w1    = (const float*)d_in[30];
  const float* c_b1    = (const float*)d_in[31];
  const float* c_w2    = (const float*)d_in[32];
  const float* c_b2    = (const float*)d_in[33];

  float* out = (float*)d_out;
  float* o_tlog = out;                            // [B,T,2]    = 512
  float* o_clog = out + 512;                      // [B,C,2]    = 2048
  float* o_qemb = out + 2560;                     // [B,Q,D]    = 2097152
  float* o_te   = out + 2560 + 2097152;           // [B,T,D]    = 262144
  float* o_ce   = o_te + 262144;                  // [B,C,D]    = 1048576

  // workspace bump allocator (~200 MB total)
  char* wsb = (char*)d_ws;
  size_t off = 0;
  auto alloc = [&](size_t nfloats) {
    float* p = (float*)(wsb + off);
    off += ((nfloats * sizeof(float) + 255) / 256) * 256;
    return p;
  };
  float* XT      = alloc((size_t)BB * TT * LTT * DD);   // 2.1M  gathered table spans
  float* XC      = alloc((size_t)BB * CC * LCC * DD);   // 16.8M gathered column spans
  float* Y1      = alloc((size_t)BB * CC * LCC * DD);   // 16.8M layer-1 output (shared)
  float* gates   = alloc((size_t)BB * CC * 2048);       // 2.1M
  float* hbuf    = alloc((size_t)BB * CC * HH);
  float* cbuf    = alloc((size_t)BB * CC * HH);
  float* pooledT = alloc((size_t)BB * TT * DD);
  float* pooledC = alloc((size_t)BB * CC * DD);
  float* te0     = alloc((size_t)BB * TT * DD);
  float* te1     = alloc((size_t)BB * TT * DD);
  float* ce0     = alloc((size_t)BB * CC * DD);
  float* qp      = alloc((size_t)BB * CC * DD);         // max nq = 1024
  float* kp      = alloc((size_t)BB * QQ * DD);         // max nkv = 2048
  float* vp      = alloc((size_t)BB * QQ * DD);
  float* attnb   = alloc((size_t)BB * CC * DD);
  float* projb   = alloc((size_t)BB * CC * DD);
  float* h1buf   = alloc((size_t)BB * CC * 256);
  (void)ws_size; (void)in_sizes; (void)n_in; (void)out_size;

  auto gemm = [&](const float* A1, int lda1, int K1,
                  const float* A2, int lda2, int K2,
                  const float* W1, const float* W2, const float* bias,
                  float* Cp, int ldc, int M, int N, int act) {
    dim3 grid(M / GBM, N / GBN);
    gemm_bf16_wmma<<<grid, 256, 0, stream>>>(A1, lda1, K1, A2, lda2, K2,
                                             W1, W2, bias, Cp, ldc, act);
  };

  // ---- gathers -----------------------------------------------------------
  gather_rows<<<BB * QQ, 256, 0, stream>>>(seq, q_ids, o_qemb, QQ);
  gather_rows<<<BB * TT * LTT, 256, 0, stream>>>(seq, t_ids, XT, TT * LTT);
  gather_rows<<<BB * CC * LCC, 256, 0, stream>>>(seq, c_ids, XC, CC * LCC);

  // ---- 2-layer BiLSTM (batched over all spans; WMMA gate GEMM per step) --
  auto bilstm = [&](float* X, int ns, int L,
                    const float* Wih, const float* Whh, const float* bvec,
                    float* pooled) {
    for (int layer = 0; layer < 2; ++layer) {
      const float* in = layer ? Y1 : X;
      float* Y = layer ? X : Y1;                   // layer-2 hs are scratch; reuse X
      for (int dir = 0; dir < 2; ++dir) {
        const float* Wi = Wih + (size_t)(layer * 2 + dir) * 2048 * DD;
        const float* Wh = Whh + (size_t)(layer * 2 + dir) * 2048 * HH;
        const float* bb = bvec + (size_t)(layer * 2 + dir) * 2048;
        hipMemsetAsync(hbuf, 0, (size_t)ns * HH * sizeof(float), stream);
        hipMemsetAsync(cbuf, 0, (size_t)ns * HH * sizeof(float), stream);
        for (int s = 0; s < L; ++s) {
          int t = dir ? (L - 1 - s) : s;
          gemm(in + (size_t)t * DD, L * DD, DD,    // x_t  (K1=1024)
               hbuf, HH, HH,                       // h    (K2=512)
               Wi, Wh, bb, gates, 2048, ns, 2048, 0);
          float* pool_ptr = (layer == 1 && s == L - 1) ? (pooled + dir * HH) : nullptr;
          lstm_point<<<(ns * HH + 255) / 256, 256, 0, stream>>>(
              gates, cbuf, hbuf, Y + (size_t)t * DD + dir * HH, (long)L * DD,
              pool_ptr, ns);
        }
      }
    }
  };
  bilstm(XT, BB * TT, LTT, tWih, tWhh, tBb, pooledT);
  bilstm(XC, BB * CC, LCC, cWih, cWhh, cBb, pooledC);

  // ---- pooling linear + leaky_relu ---------------------------------------
  gemm(pooledT, DD, DD, nullptr, 0, 0, t_pw, nullptr, t_pb, te0, DD, BB * TT, DD, 1);
  gemm(pooledC, DD, DD, nullptr, 0, 0, c_pw, nullptr, c_pb, ce0, DD, BB * CC, DD, 1);

  // ---- MHA blocks ---------------------------------------------------------
  auto mha_run = [&](const float* qsrc, int nq, const float* kvsrc, int nkv,
                     const float* in_w, const float* in_b,
                     const float* out_w, const float* out_b,
                     int qdiv, int kmul, int nk,
                     const float* resid, float* dst) {
    gemm(qsrc, DD, DD, nullptr, 0, 0, in_w, nullptr, in_b, qp, DD, nq, DD, 0);
    gemm(kvsrc, DD, DD, nullptr, 0, 0, in_w + (size_t)DD * DD, nullptr, in_b + DD,
         kp, DD, nkv, DD, 0);
    gemm(kvsrc, DD, DD, nullptr, 0, 0, in_w + (size_t)2 * DD * DD, nullptr, in_b + 2 * DD,
         vp, DD, nkv, DD, 0);
    attn_kernel<<<nq * NHEAD, DHEAD, 0, stream>>>(qp, kp, vp, attnb, qdiv, kmul, nk);
    gemm(attnb, DD, DD, nullptr, 0, 0, out_w, nullptr, out_b, projb, DD, nq, DD, 0);
    resid_l2norm<<<nq, 256, 0, stream>>>(resid, projb, dst);
  };

  // table <- its 4 columns; then table <- question; column <- question
  mha_run(te0, BB * TT, ce0, BB * CC, tc_iw, tc_ib, tc_ow, tc_ob, 1, 4, 4, te0, te1);
  mha_run(te1, BB * TT, o_qemb, BB * QQ, tq_iw, tq_ib, tq_ow, tq_ob, TT, QQ, QQ, te1, o_te);
  mha_run(ce0, BB * CC, o_qemb, BB * QQ, cq_iw, cq_ib, cq_ow, cq_ob, CC, QQ, QQ, ce0, o_ce);

  // ---- classifier heads ---------------------------------------------------
  gemm(o_te, DD, DD, nullptr, 0, 0, t_w1, nullptr, t_b1, h1buf, 256, BB * TT, 256, 1);
  cls2_kernel<<<BB * TT, 256, 0, stream>>>(h1buf, t_w2, t_b2, o_tlog);
  gemm(o_ce, DD, DD, nullptr, 0, 0, c_w1, nullptr, c_b1, h1buf, 256, BB * CC, 256, 1);
  cls2_kernel<<<BB * CC, 256, 0, stream>>>(h1buf, c_w2, c_b2, o_clog);
}
